// HSIC_41008347742755
// MI455X (gfx1250) — matrix operand order repaired and checked by reading, not compiled, not run on hardware
//
#include <hip/hip_runtime.h>

// ---------------------------------------------------------------------------
// HSIC on MI455X (gfx1250), fully fused:
//   result = [ sum(K.*L) - (2/n) sum_i rK_i*rL_i + sK*sL/n^2 ] / n^2
// where K = exp(-0.5*d2(X)), L = exp(-0.5*d2(Y)), rK/rL row sums.
// Gram matrices X*X^T and Y*Y^T computed with v_wmma_f32_16x16x32_f16.
// All data (8 MB as f16) is L2-resident; kernel is matrix-core bound.
// Diagonal handled like the reference: d2 clamped at 0 (no special case);
// squared norms are computed from the f16-rounded rows so d2_ii ~ 0.
// ---------------------------------------------------------------------------

typedef __attribute__((ext_vector_type(16))) _Float16 v16h;
typedef __attribute__((ext_vector_type(8)))  _Float16 v8h;
typedef __attribute__((ext_vector_type(8)))  float    v8f;

#define HSIC_N 4096
#define HSIC_D 512

// workspace layout (bytes)
#define XH_OFF   0u
#define YH_OFF   (XH_OFF + (unsigned)HSIC_N * HSIC_D * 2u)      // 4 MiB
#define SQX_OFF  (YH_OFF + (unsigned)HSIC_N * HSIC_D * 2u)      // 8 MiB
#define SQY_OFF  (SQX_OFF + (unsigned)HSIC_N * 4u)
#define RKP_OFF  (SQY_OFF + (unsigned)HSIC_N * 4u)              // [32][4096] f32
#define RLP_OFF  (RKP_OFF + 32u * HSIC_N * 4u)
#define PRD_OFF  (RLP_OFF + 32u * HSIC_N * 4u)                  // [1024] f32

static __device__ inline v8f vzero8() {
    v8f z;
#pragma unroll
    for (int i = 0; i < 8; ++i) z[i] = 0.0f;
    return z;
}

// ---------------------------------------------------------------------------
// Prep: f32 -> f16 copies + per-row squared norms computed from the ROUNDED
// (f16) values so that d2_ii cancels consistently with the WMMA dot products.
// ---------------------------------------------------------------------------
__global__ __launch_bounds__(256) void hsic_prep(
    const float* __restrict__ X, const float* __restrict__ Y,
    _Float16* __restrict__ Xh, _Float16* __restrict__ Yh,
    float* __restrict__ sqX, float* __restrict__ sqY)
{
    __shared__ float red[256];
    const int row = blockIdx.x;
    const int tid = threadIdx.x;

    float sum = 0.0f;
    for (int c = tid; c < HSIC_D; c += 256) {
        float x = X[(size_t)row * HSIC_D + c];
        _Float16 h = (_Float16)x;
        Xh[(size_t)row * HSIC_D + c] = h;
        float v = (float)h;
        sum += v * v;
    }
    red[tid] = sum;
    __syncthreads();
    for (int off = 128; off > 0; off >>= 1) {
        if (tid < off) red[tid] += red[tid + off];
        __syncthreads();
    }
    if (tid == 0) sqX[row] = red[0];
    __syncthreads();

    sum = 0.0f;
    for (int c = tid; c < HSIC_D; c += 256) {
        float y = Y[(size_t)row * HSIC_D + c];
        _Float16 h = (_Float16)y;
        Yh[(size_t)row * HSIC_D + c] = h;
        float v = (float)h;
        sum += v * v;
    }
    red[tid] = sum;
    __syncthreads();
    for (int off = 128; off > 0; off >>= 1) {
        if (tid < off) red[tid] += red[tid + off];
        __syncthreads();
    }
    if (tid == 0) sqY[row] = red[0];
}

// ---------------------------------------------------------------------------
// Main: one 128x128 tile of (i,j) per block; 8 waves, each wave a 32x64
// sub-tile = 2x4 WMMA fragments per matrix.  Pass 1: K tile (from X),
// pass 2: L tile (from Y) fused with exp / K.*L / row-sum partials.
// Fragment loads come straight from the L2-resident f16 arrays following
// a contiguous k = 16*half + s order for both A (16x32) and B (32x16).
// ---------------------------------------------------------------------------
__global__ __launch_bounds__(256) void hsic_main(
    const _Float16* __restrict__ Xh, const _Float16* __restrict__ Yh,
    const float* __restrict__ sqX, const float* __restrict__ sqY,
    float* __restrict__ rKp, float* __restrict__ rLp,
    float* __restrict__ prodArr)
{
    __shared__ float srowK[128];
    __shared__ float srowL[128];
    __shared__ float sprod[8];

    const int tid  = threadIdx.x;
    const int lane = tid & 31;
    const int w    = tid >> 5;     // wave 0..7
    const int wr   = w >> 1;       // wave row 0..3
    const int wc   = w & 1;        // wave col 0..1
    const int half = lane >> 4;    // 0 or 1 (lane group)
    const int ln   = lane & 15;

    const int ib = blockIdx.y * 128;
    const int jb = blockIdx.x * 128;
    const int i0 = ib + wr * 32;   // wave's 32 rows
    const int j0 = jb + wc * 64;   // wave's 64 cols

    if (tid < 128) { srowK[tid] = 0.0f; srowL[tid] = 0.0f; }
    __syncthreads();

    v8f acc[2][4];
    v8f Kf[2][4];
    float sqrA[2][8];
    float sqcB[4];

    // ---------------- pass 1: Gram(X) -> K tile ----------------
#pragma unroll
    for (int mt = 0; mt < 2; ++mt)
#pragma unroll
        for (int nt = 0; nt < 4; ++nt) acc[mt][nt] = vzero8();

    for (int k0 = 0; k0 < HSIC_D; k0 += 32) {
        v16h a[2], b[4];
#pragma unroll
        for (int mt = 0; mt < 2; ++mt) {
            const _Float16* p = Xh + (size_t)(i0 + mt * 16 + ln) * HSIC_D + k0;
            v8h lo = *(const v8h*)(p + 16 * half);       // K = 16*half .. +8
            v8h hi = *(const v8h*)(p + 16 * half + 8);   // K = 16*half+8 .. +8
            a[mt] = __builtin_shufflevector(lo, hi, 0,1,2,3,4,5,6,7,8,9,10,11,12,13,14,15);
        }
#pragma unroll
        for (int nt = 0; nt < 4; ++nt) {
            const _Float16* p = Xh + (size_t)(j0 + nt * 16 + ln) * HSIC_D + k0 + 16 * half;
            b[nt] = *(const v16h*)p;                     // K = 16*half .. +16
        }
#pragma unroll
        for (int mt = 0; mt < 2; ++mt)
#pragma unroll
            for (int nt = 0; nt < 4; ++nt)
                acc[mt][nt] = __builtin_amdgcn_wmma_f32_16x16x32_f16(
                    false, a[mt], false, b[nt], (short)0, acc[mt][nt], false, false);
    }

#pragma unroll
    for (int mt = 0; mt < 2; ++mt)
#pragma unroll
        for (int r = 0; r < 8; ++r)
            sqrA[mt][r] = sqX[i0 + mt * 16 + 8 * half + r];
#pragma unroll
    for (int nt = 0; nt < 4; ++nt) sqcB[nt] = sqX[j0 + nt * 16 + ln];

#pragma unroll
    for (int mt = 0; mt < 2; ++mt)
#pragma unroll
        for (int nt = 0; nt < 4; ++nt)
#pragma unroll
            for (int r = 0; r < 8; ++r) {
                float d2 = sqrA[mt][r] + sqcB[nt] - 2.0f * acc[mt][nt][r];
                d2 = fmaxf(d2, 0.0f);                    // branch-free clamp
                Kf[mt][nt][r] = __expf(-0.5f * d2);
            }

    // ---------------- pass 2: Gram(Y) -> L tile, fuse ----------------
#pragma unroll
    for (int mt = 0; mt < 2; ++mt)
#pragma unroll
        for (int nt = 0; nt < 4; ++nt) acc[mt][nt] = vzero8();

    for (int k0 = 0; k0 < HSIC_D; k0 += 32) {
        v16h a[2], b[4];
#pragma unroll
        for (int mt = 0; mt < 2; ++mt) {
            const _Float16* p = Yh + (size_t)(i0 + mt * 16 + ln) * HSIC_D + k0;
            v8h lo = *(const v8h*)(p + 16 * half);
            v8h hi = *(const v8h*)(p + 16 * half + 8);
            a[mt] = __builtin_shufflevector(lo, hi, 0,1,2,3,4,5,6,7,8,9,10,11,12,13,14,15);
        }
#pragma unroll
        for (int nt = 0; nt < 4; ++nt) {
            const _Float16* p = Yh + (size_t)(j0 + nt * 16 + ln) * HSIC_D + k0 + 16 * half;
            b[nt] = *(const v16h*)p;
        }
#pragma unroll
        for (int mt = 0; mt < 2; ++mt)
#pragma unroll
            for (int nt = 0; nt < 4; ++nt)
                acc[mt][nt] = __builtin_amdgcn_wmma_f32_16x16x32_f16(
                    false, a[mt], false, b[nt], (short)0, acc[mt][nt], false, false);
    }

#pragma unroll
    for (int mt = 0; mt < 2; ++mt)
#pragma unroll
        for (int r = 0; r < 8; ++r)
            sqrA[mt][r] = sqY[i0 + mt * 16 + 8 * half + r];
#pragma unroll
    for (int nt = 0; nt < 4; ++nt) sqcB[nt] = sqY[j0 + nt * 16 + ln];

    float prodloc = 0.0f;
    v8f rowKs[2], rowLs[2];
#pragma unroll
    for (int mt = 0; mt < 2; ++mt) { rowKs[mt] = vzero8(); rowLs[mt] = vzero8(); }

#pragma unroll
    for (int mt = 0; mt < 2; ++mt)
#pragma unroll
        for (int nt = 0; nt < 4; ++nt)
#pragma unroll
            for (int r = 0; r < 8; ++r) {
                float d2 = sqrA[mt][r] + sqcB[nt] - 2.0f * acc[mt][nt][r];
                d2 = fmaxf(d2, 0.0f);
                const float Lv = __expf(-0.5f * d2);
                const float Kv = Kf[mt][nt][r];
                prodloc += Kv * Lv;
                rowKs[mt][r] += Kv;
                rowLs[mt][r] += Lv;
            }

    // ---- wave reductions: row sums over the wave's 64 columns ----
#pragma unroll
    for (int mt = 0; mt < 2; ++mt)
#pragma unroll
        for (int r = 0; r < 8; ++r) {
            float kv = rowKs[mt][r];
            float lv = rowLs[mt][r];
#pragma unroll
            for (int s = 1; s < 16; s <<= 1) {  // reduce within each 16-lane half
                kv += __shfl_xor(kv, s, 32);
                lv += __shfl_xor(lv, s, 32);
            }
            if (ln == 0) {
                const int lrow = wr * 32 + mt * 16 + 8 * half + r;
                atomicAdd(&srowK[lrow], kv);  // ds_add_f32, 2 waves contend
                atomicAdd(&srowL[lrow], lv);
            }
        }

#pragma unroll
    for (int s = 1; s < 32; s <<= 1) prodloc += __shfl_xor(prodloc, s, 32);
    if (lane == 0) sprod[w] = prodloc;
    __syncthreads();

    // deterministic per-block partial outputs (no global atomics)
    if (tid < 128) {
        rKp[(size_t)blockIdx.x * HSIC_N + ib + tid] = srowK[tid];
        rLp[(size_t)blockIdx.x * HSIC_N + ib + tid] = srowL[tid];
    }
    if (tid == 0) {
        float p = 0.0f;
#pragma unroll
        for (int q = 0; q < 8; ++q) p += sprod[q];
        prodArr[blockIdx.y * 32 + blockIdx.x] = p;
    }
}

// ---------------------------------------------------------------------------
// Final: fold 32 j-block partials per row, compute s, t, dot, P, emit scalar.
// ---------------------------------------------------------------------------
__global__ __launch_bounds__(256) void hsic_final(
    const float* __restrict__ rKp, const float* __restrict__ rLp,
    const float* __restrict__ prodArr, float* __restrict__ out)
{
    __shared__ float sd[256], ssm[256], stm[256], sp[256];
    const int tid = threadIdx.x;
    float dot = 0.0f, s = 0.0f, t = 0.0f, P = 0.0f;
    for (int i = tid; i < HSIC_N; i += 256) {
        float rk = 0.0f, rl = 0.0f;
        for (int b = 0; b < 32; ++b) {
            rk += rKp[(size_t)b * HSIC_N + i];
            rl += rLp[(size_t)b * HSIC_N + i];
        }
        dot += rk * rl; s += rk; t += rl;
    }
    for (int i = tid; i < 1024; i += 256) P += prodArr[i];
    sd[tid] = dot; ssm[tid] = s; stm[tid] = t; sp[tid] = P;
    __syncthreads();
    for (int off = 128; off > 0; off >>= 1) {
        if (tid < off) {
            sd[tid] += sd[tid + off];
            ssm[tid] += ssm[tid + off];
            stm[tid] += stm[tid + off];
            sp[tid] += sp[tid + off];
        }
        __syncthreads();
    }
    if (tid == 0) {
        const float n = (float)HSIC_N;
        const float val = sp[0] - (2.0f / n) * sd[0] + (ssm[0] / n) * (stm[0] / n);
        out[0] = val / (n * n);
    }
}

// ---------------------------------------------------------------------------
extern "C" void kernel_launch(void* const* d_in, const int* in_sizes, int n_in,
                              void* d_out, int out_size, void* d_ws, size_t ws_size,
                              hipStream_t stream) {
    (void)in_sizes; (void)n_in; (void)out_size; (void)ws_size;
    const float* X = (const float*)d_in[0];
    const float* Y = (const float*)d_in[1];
    float* out = (float*)d_out;

    char* ws = (char*)d_ws;
    _Float16* Xh = (_Float16*)(ws + XH_OFF);
    _Float16* Yh = (_Float16*)(ws + YH_OFF);
    float* sqX = (float*)(ws + SQX_OFF);
    float* sqY = (float*)(ws + SQY_OFF);
    float* rKp = (float*)(ws + RKP_OFF);
    float* rLp = (float*)(ws + RLP_OFF);
    float* prodArr = (float*)(ws + PRD_OFF);

    hsic_prep<<<HSIC_N, 256, 0, stream>>>(X, Y, Xh, Yh, sqX, sqY);

    dim3 grid(32, 32);
    hsic_main<<<grid, 256, 0, stream>>>(Xh, Yh, sqX, sqY, rKp, rLp, prodArr);

    hsic_final<<<1, 256, 0, stream>>>(rKp, rLp, prodArr, out);
}
